// moe_3135326126535
// MI455X (gfx1250) — matrix operand order repaired and checked by reading, not compiled, not run on hardware
//
#include <hip/hip_runtime.h>
#include <hip/hip_bf16.h>
#include <math.h>

typedef __attribute__((ext_vector_type(16))) _Float16 v16h;
typedef __attribute__((ext_vector_type(8)))  _Float16 v8h;
typedef __attribute__((ext_vector_type(8)))  float    v8f;

#define NN   10000
#define DD   128
#define DEG  32
#define EE   (NN*DEG)
#define NEXP 8
#define HH   128

union V16U { v16h v; v8h h[2]; };

// ---------- prep: convert node features to f16 ----------
__global__ void k_prep_x(const float* __restrict__ x, _Float16* __restrict__ x16, int n) {
    int i = blockIdx.x * 256 + threadIdx.x;
    if (i < n) x16[i] = (_Float16)x[i];
}

// ---------- prep: transpose W1[:, 0:256, :] -> (e, n, k) f16 ; keep attr row f32 ----------
__global__ void k_prep_w1(const float* __restrict__ W1, _Float16* __restrict__ w1t,
                          float* __restrict__ w1last) {
    int i = blockIdx.x * 256 + threadIdx.x;         // e*128*256 + n*256 + k
    if (i < NEXP * HH * 256) {
        int k = i & 255;
        int n = (i >> 8) & 127;
        int e = i >> 15;
        w1t[i] = (_Float16)W1[((size_t)e * 257 + k) * HH + n];
    }
    if (i < NEXP * HH) {
        int n = i & 127;
        int e = i >> 7;
        w1last[i] = W1[((size_t)e * 257 + 256) * HH + n];
    }
}

// ---------- zero small accumulators ----------
__global__ void k_zero(float* p, int n) {
    int i = blockIdx.x * 32 + threadIdx.x;
    if (i < n) p[i] = 0.f;
}

// ---------- gating: logits, top-2 softmax, load/relevance accumulation ----------
__global__ void k_gate(const float* __restrict__ x, const float* __restrict__ wg,
                       int* __restrict__ e0a, int* __restrict__ e1a,
                       float* __restrict__ g0a, float* __restrict__ g1a,
                       float* __restrict__ rel, float* __restrict__ loadv) {
    int node = blockIdx.x * 8 + (threadIdx.x >> 5);
    int lane = threadIdx.x & 31;
    if (node >= NN) return;
    float p[NEXP];
#pragma unroll
    for (int e = 0; e < NEXP; e++) p[e] = 0.f;
    const float* xr = x + (size_t)node * DD;
    for (int d = lane; d < DD; d += 32) {
        float xv = xr[d];
#pragma unroll
        for (int e = 0; e < NEXP; e++) p[e] += xv * wg[d * NEXP + e];
    }
#pragma unroll
    for (int off = 16; off; off >>= 1) {
#pragma unroll
        for (int e = 0; e < NEXP; e++) p[e] += __shfl_xor(p[e], off, 32);
    }
    if (lane == 0) {
        int b0 = 0; float v0 = p[0];
        for (int e = 1; e < NEXP; e++) if (p[e] > v0) { v0 = p[e]; b0 = e; }
        int b1 = -1; float v1 = -INFINITY;
        for (int e = 0; e < NEXP; e++) if (e != b0 && p[e] > v1) { v1 = p[e]; b1 = e; }
        float g0 = 1.0f / (1.0f + expf(v1 - v0));   // softmax over {v0,v1}
        float g1 = 1.0f - g0;
        e0a[node] = b0; e1a[node] = b1; g0a[node] = g0; g1a[node] = g1;
        atomicAdd(&rel[b0], g0);   atomicAdd(&rel[b1], g1);
        atomicAdd(&loadv[b0], 1.f); atomicAdd(&loadv[b1], 1.f);
    }
}

// ---------- loss scalar ----------
__global__ void k_loss(const float* __restrict__ rel, const float* __restrict__ loadv,
                       float* __restrict__ out_loss) {
    if (threadIdx.x == 0 && blockIdx.x == 0) {
        float cv[2];
        const float* arr[2] = { rel, loadv };
        for (int t = 0; t < 2; t++) {
            float m = 0.f;
            for (int e = 0; e < NEXP; e++) m += arr[t][e];
            m *= (1.0f / NEXP);
            float v = 0.f;
            for (int e = 0; e < NEXP; e++) { float d = arr[t][e] - m; v += d * d; }
            v *= (1.0f / (NEXP - 1));                 // ddof=1
            cv[t] = v / (m * m + 1e-10f);
        }
        *out_loss = 0.01f * (cv[0] + cv[1]);
    }
}

// ---------- fused expert-MLP (WMMA) + per-node threshold/mask ----------
// one block per node; 2 waves = 2 expert slots; each wave owns all 32 edges
// (two 16-row M tiles) so every B (weight) tile load feeds TWO WMMAs.
__global__ void __launch_bounds__(64, 1)
k_mlp(const _Float16* __restrict__ X16,
      const _Float16* __restrict__ W1T,    // [NE][128][256] n-major
      const float* __restrict__ W1LAST,    // [NE][128] attr row
      const float* __restrict__ B1,        // [NE][128]
      const float* __restrict__ W2,        // [NE][128]
      const float* __restrict__ B2,        // [NE]
      const int* __restrict__ DST,         // [E]
      const float* __restrict__ ATTR,      // [E]
      const float* __restrict__ TEMP,      // [1]
      const int* __restrict__ e0a, const int* __restrict__ e1a,
      const float* __restrict__ g0a, const float* __restrict__ g1a,
      float* __restrict__ out_mask)        // d_out[0..E)
{
    __shared__ float sgate[2][DEG];

    const int node = blockIdx.x;
    const int wave = threadIdx.x >> 5;     // == expert slot
    const int lane = threadIdx.x & 31;
    const int slot = wave;
    const int hl = lane & 15, hh = lane >> 4;

    const int   e = (slot == 0) ? e0a[node] : e1a[node];
    const float g = (slot == 0) ? g0a[node] : g1a[node];
    const int edgeBase = node * DEG;

    // src row identical for all 32 edges of this node
    const _Float16* xsrc = X16 + (size_t)node * DD;
    const _Float16* xd0  = X16 + (size_t)DST[edgeBase      + hl] * DD;  // tile0: m = hl
    const _Float16* xd1  = X16 + (size_t)DST[edgeBase + 16 + hl] * DD;  // tile1: m = 16+hl
    const _Float16* wt   = W1T + ((size_t)e * HH + hl) * 256;           // B column n = hl (+nt*16)
    const int koffA = hh * 8;    // A: lanes 0-15 hold K {0..7,16..23}; lanes 16-31 hold K {8..15,24..31}
    const int koffB = hh * 8;    // B: lanes 0-15 hold K {0..7,16..23}; lanes 16-31 hold K {8..15,24..31}

    v8f zero = {};
    v8f acc0[8], acc1[8];
#pragma unroll
    for (int nt = 0; nt < 8; nt++) { acc0[nt] = zero; acc1[nt] = zero; }

#pragma unroll
    for (int kc = 0; kc < 8; kc++) {                 // K = kc*32 (src K 0..127, dst K 128..255)
        V16U a0, a1;
        if (kc < 4) {                                // src half: both M tiles share A exactly
            const _Float16* ab = xsrc + kc * 32;
            a0.h[0] = *(const v8h*)(ab + koffA);
            a0.h[1] = *(const v8h*)(ab + 16 + koffA);
            a1 = a0;
        } else {                                     // dst half: per-tile gathers
            const _Float16* ab0 = xd0 + (kc - 4) * 32;
            const _Float16* ab1 = xd1 + (kc - 4) * 32;
            a0.h[0] = *(const v8h*)(ab0 + koffA);
            a0.h[1] = *(const v8h*)(ab0 + 16 + koffA);
            a1.h[0] = *(const v8h*)(ab1 + koffA);
            a1.h[1] = *(const v8h*)(ab1 + 16 + koffA);
        }
#pragma unroll
        for (int nt = 0; nt < 8; nt++) {
            const _Float16* wb = wt + (size_t)nt * (16 * 256) + kc * 32 + koffB;
            V16U b;
            b.h[0] = *(const v8h*)(wb);
            b.h[1] = *(const v8h*)(wb + 16);
            acc0[nt] = __builtin_amdgcn_wmma_f32_16x16x32_f16(
                false, a0.v, false, b.v, (short)0, acc0[nt], false, false);
            acc1[nt] = __builtin_amdgcn_wmma_f32_16x16x32_f16(
                false, a1.v, false, b.v, (short)0, acc1[nt], false, false);
        }
    }

    // rank-1 attr term + bias + relu + dot with W2, for both M tiles
    float att0[8], att1[8];
#pragma unroll
    for (int j = 0; j < 8; j++) {                    // C: m = j + 8*hh (within tile)
        att0[j] = ATTR[edgeBase      + hh * 8 + j];
        att1[j] = ATTR[edgeBase + 16 + hh * 8 + j];
    }

    float z0[8], z1[8];
#pragma unroll
    for (int j = 0; j < 8; j++) { z0[j] = 0.f; z1[j] = 0.f; }
#pragma unroll
    for (int nt = 0; nt < 8; nt++) {
        const int n = hl + nt * 16;                  // C: n = hl
        const float wl  = W1LAST[e * HH + n];
        const float bb  = B1[e * HH + n];
        const float w2v = W2[e * HH + n];
#pragma unroll
        for (int j = 0; j < 8; j++) {
            float h0 = acc0[nt][j] + att0[j] * wl + bb;
            h0 = h0 > 0.f ? h0 : 0.f;
            z0[j] += h0 * w2v;
            float h1 = acc1[nt][j] + att1[j] * wl + bb;
            h1 = h1 > 0.f ? h1 : 0.f;
            z1[j] += h1 * w2v;
        }
    }
    // reduce over the 16 n-lanes (xor masks stay inside each half-wave)
#pragma unroll
    for (int off = 1; off < 16; off <<= 1) {
#pragma unroll
        for (int j = 0; j < 8; j++) {
            z0[j] += __shfl_xor(z0[j], off, 32);
            z1[j] += __shfl_xor(z1[j], off, 32);
        }
    }

    const float invT = 1.0f / TEMP[0];
    const float b2v  = B2[e];
    if (hl == 0) {                                   // lane 0 -> m 0..7, lane 16 -> m 8..15
#pragma unroll
        for (int j = 0; j < 8; j++) {
            float za = (z0[j] + b2v) * invT;
            float zb = (z1[j] + b2v) * invT;
            sgate[slot][     hh * 8 + j] = g * (1.0f / (1.0f + expf(-za)));
            sgate[slot][16 + hh * 8 + j] = g * (1.0f / (1.0f + expf(-zb)));
        }
    }
    __syncthreads();

    if (wave == 0) {
        // gated = mean over 8 experts = (g0*o0 + g1*o1)/8
        float v = (sgate[0][lane] + sgate[1][lane]) * 0.125f;

        // r = round(32 * (g0*k[e0] + g1*k[e1])), clamped to [1,32]  (counts == 32 for all nodes)
        const float selk = g0a[node] * (float)(e0a[node] + 1) * 0.1f
                         + g1a[node] * (float)(e1a[node] + 1) * 0.1f;
        int r = (int)rintf(selk * (float)DEG);
        r = r < 1 ? 1 : (r > DEG ? DEG : r);

        int cnt_gt = 0, cnt_ge = 0;
        for (int j = 0; j < DEG; j++) {
            float vj = __shfl(v, j, 32);
            cnt_gt += (vj > v);
            cnt_ge += (vj >= v);
        }
        bool isTh = (cnt_gt <= r - 1) && (r - 1 < cnt_ge);   // value at descending rank r-1
        unsigned long long bal = __ballot(isTh);
        int sl = __ffsll(bal) - 1;
        float thresh = __shfl(v, sl, 32);
        out_mask[node * DEG + lane] = (v - thresh + 1e-12f > 0.f) ? 1.f : 0.f;
    }
}

extern "C" void kernel_launch(void* const* d_in, const int* in_sizes, int n_in,
                              void* d_out, int out_size, void* d_ws, size_t ws_size,
                              hipStream_t stream) {
    const float* input_nodes = (const float*)d_in[0];
    const int*   adj         = (const int*)d_in[1];     // [2][E]: src then dst
    const float* attr        = (const float*)d_in[2];
    const float* temp        = (const float*)d_in[3];
    const float* w_gate      = (const float*)d_in[4];
    const float* W1          = (const float*)d_in[5];
    const float* b1          = (const float*)d_in[6];
    const float* W2          = (const float*)d_in[7];
    const float* b2          = (const float*)d_in[8];
    (void)in_sizes; (void)n_in; (void)out_size; (void)ws_size;

    const int* dst = adj + EE;

    // workspace carve-up (all 16B aligned)
    char* ws = (char*)d_ws;
    _Float16* X16    = (_Float16*)ws;            ws += (size_t)NN * DD * 2;         // 2,560,000 B
    _Float16* W1T    = (_Float16*)ws;            ws += (size_t)NEXP * HH * 256 * 2; // 524,288 B
    float*    W1LAST = (float*)ws;               ws += (size_t)NEXP * HH * 4;       // 4,096 B
    float*    rel    = (float*)ws;               ws += 8 * 4;
    float*    loadv  = (float*)ws;               ws += 8 * 4;
    int*      e0a    = (int*)ws;                 ws += (size_t)NN * 4;
    int*      e1a    = (int*)ws;                 ws += (size_t)NN * 4;
    float*    g0a    = (float*)ws;               ws += (size_t)NN * 4;
    float*    g1a    = (float*)ws;               ws += (size_t)NN * 4;

    float* out_mask = (float*)d_out;
    float* out_loss = out_mask + EE;

    k_prep_x <<<(NN * DD + 255) / 256, 256, 0, stream>>>(input_nodes, X16, NN * DD);
    k_prep_w1<<<(NEXP * HH * 256 + 255) / 256, 256, 0, stream>>>(W1, W1T, W1LAST);
    k_zero   <<<1, 32, 0, stream>>>(rel, 16);   // rel + loadv are contiguous
    k_gate   <<<(NN + 7) / 8, 256, 0, stream>>>(input_nodes, w_gate, e0a, e1a, g0a, g1a, rel, loadv);
    k_loss   <<<1, 32, 0, stream>>>(rel, loadv, out_loss);
    k_mlp    <<<NN, 64, 0, stream>>>(X16, W1T, W1LAST, b1, W2, b2,
                                     dst, attr, temp, e0a, e1a, g0a, g1a, out_mask);
}
